// BimodalCrossAttentionBlock_50199577756230
// MI455X (gfx1250) — matrix operand order repaired and checked by reading, not compiled, not run on hardware
//
#include <hip/hip_runtime.h>
#include <hip/hip_bf16.h>

// ---------------------------------------------------------------------------
// Types for gfx1250 WMMA (wave32)
// ---------------------------------------------------------------------------
typedef __attribute__((ext_vector_type(16))) __bf16       v16bf;
typedef __attribute__((ext_vector_type(8)))  float        v8f;
typedef __attribute__((ext_vector_type(4)))  unsigned int u32x4;
typedef __attribute__((ext_vector_type(4)))  __bf16       bf16x4;

union FragBF {
    v16bf bf;
    u32x4 u[2];
};

enum { EP_BF16 = 0, EP_GELU_BF16 = 1, EP_RES_F32 = 2 };

// Block tile 128x256, K-step 32. 8 wave32 in a 2x4 grid, 64x64 per wave.
#define BM 128
#define BN 256
#define BK 32
#define LDSK 40                    // BK + 8 pad: 80-byte row stride, b128-aligned
#define A_ELEMS (BM * LDSK)        // 5120 bf16 = 10240 B
#define B_ELEMS (BN * LDSK)        // 10240 bf16 = 20480 B
#define BUF_ELEMS (A_ELEMS + B_ELEMS)

// ---------------------------------------------------------------------------
// CDNA5 async global->LDS copy (ASYNCcnt) — 16 bytes per lane.
// VDST operand = LDS byte offset; generic LDS pointer low 32 bits are the
// in-aperture LDS offset per the flat-addressing rules.
// ---------------------------------------------------------------------------
__device__ __forceinline__ unsigned lds_off(const void* p)
{
    return (unsigned)(uintptr_t)p;
}

__device__ __forceinline__ void async_load_b128(const void* gptr, unsigned ldsoff)
{
    asm volatile("global_load_async_to_lds_b128 %0, %1, off"
                 :: "v"(ldsoff), "v"(gptr)
                 : "memory");
}

__device__ __forceinline__ void wait_async0()
{
    asm volatile("s_wait_asynccnt 0x0" ::: "memory");
}

// ---------------------------------------------------------------------------
// fp32 -> bf16 conversion, n multiple of 4
// ---------------------------------------------------------------------------
__global__ __launch_bounds__(256)
void cvt_f32_bf16(const float* __restrict__ src, __bf16* __restrict__ dst, long n4)
{
    long i = (long)blockIdx.x * blockDim.x + threadIdx.x;
    if (i >= n4) return;
    long idx = i * 4;
    const float4 v = *(const float4*)(src + idx);
    bf16x4 o;
    o.x = (__bf16)v.x; o.y = (__bf16)v.y; o.z = (__bf16)v.z; o.w = (__bf16)v.w;
    *(bf16x4*)(dst + idx) = o;
}

// ---------------------------------------------------------------------------
// Tiled bf16 WMMA GEMM:  C[M,N] = A[M,K] @ W[N,K]^T + bias (+ epilogue)
// Double-buffered LDS fed by async global->LDS loads.
// Per K-step schedule: s_wait_asynccnt 0 -> barrier -> issue next tile -> WMMA.
// EP_RES_F32 may have Cv alias res (same index read-then-write per thread).
// ---------------------------------------------------------------------------
template <int EPI>
__global__ __launch_bounds__(256)
void gemm_bf16_wmma(const __bf16* __restrict__ A,
                    const __bf16* __restrict__ W,
                    const float* __restrict__ bias,
                    const float* __restrict__ res,
                    void* __restrict__ Cv,
                    int M, int N, int K)
{
    __shared__ __attribute__((aligned(16))) __bf16 lds[2 * BUF_ELEMS];

    const int tid  = threadIdx.x;
    const int lane = tid & 31;
    const int wid  = tid >> 5;
    const int wm   = wid >> 2;     // 0..1  (64 rows each)
    const int wn   = wid & 3;      // 0..3  (64 cols each)
    const int bm   = blockIdx.y * BM;
    const int bn   = blockIdx.x * BN;
    const int half = lane >> 4;    // K-half selector (this code's contiguous k = 16*half + s order)
    const int l16  = lane & 15;

    v8f acc[4][4];
#pragma unroll
    for (int mi = 0; mi < 4; ++mi)
#pragma unroll
        for (int ni = 0; ni < 4; ++ni)
            acc[mi][ni] = (v8f){0.f, 0.f, 0.f, 0.f, 0.f, 0.f, 0.f, 0.f};

    // ---- async stage of one K-tile into LDS buffer `buf` ----
    auto stage = [&](int k0, int buf) {
        __bf16* sA = lds + buf * BUF_ELEMS;
        __bf16* sB = sA + A_ELEMS;
#pragma unroll
        for (int i = 0; i < 2; ++i) {            // A: 128 rows x 32 bf16
            int g = tid + i * 256;
            int row = g >> 2, col = (g & 3) * 8;
            async_load_b128(A + (size_t)(bm + row) * K + k0 + col,
                            lds_off(&sA[row * LDSK + col]));
        }
#pragma unroll
        for (int i = 0; i < 4; ++i) {            // W: 256 rows x 32 bf16
            int g = tid + i * 256;
            int row = g >> 2, col = (g & 3) * 8;
            async_load_b128(W + (size_t)(bn + row) * K + k0 + col,
                            lds_off(&sB[row * LDSK + col]));
        }
    };

    stage(0, 0);
    const int nk = K / BK;

    for (int ki = 0; ki < nk; ++ki) {
        wait_async0();        // own async writes for tile ki are in LDS
        __syncthreads();      // everyone's writes for tile ki are in LDS
        if (ki + 1 < nk)
            stage((ki + 1) * BK, (ki + 1) & 1);  // overlaps with WMMA below

        const __bf16* cA = lds + (ki & 1) * BUF_ELEMS;
        const __bf16* cB = cA + A_ELEMS;

        FragBF bfrag[4];
#pragma unroll
        for (int ni = 0; ni < 4; ++ni) {
            const __bf16* base = &cB[(wn * 64 + ni * 16 + l16) * LDSK + 16 * half];
            bfrag[ni].u[0] = *(const u32x4*)(base);
            bfrag[ni].u[1] = *(const u32x4*)(base + 8);
        }
#pragma unroll
        for (int mi = 0; mi < 4; ++mi) {
            FragBF afrag;
            const __bf16* base = &cA[(wm * 64 + mi * 16 + l16) * LDSK + 16 * half];
            afrag.u[0] = *(const u32x4*)(base);        // K = 0..7   (+16*half)
            afrag.u[1] = *(const u32x4*)(base + 8);    // K = 8..15  (+16*half)
#pragma unroll
            for (int ni = 0; ni < 4; ++ni) {
                acc[mi][ni] = __builtin_amdgcn_wmma_f32_16x16x32_bf16(
                    false, afrag.bf, false, bfrag[ni].bf,
                    (short)0, acc[mi][ni], false, false);
            }
        }
    }

    // ---- epilogue: C layout VGPR r -> M = r + 8*half, lane -> N = l16 ----
#pragma unroll
    for (int mi = 0; mi < 4; ++mi) {
        const int row0 = bm + wm * 64 + mi * 16 + 8 * half;
#pragma unroll
        for (int ni = 0; ni < 4; ++ni) {
            const int col = bn + wn * 64 + ni * 16 + l16;
            const float bv = bias[col];
#pragma unroll
            for (int r = 0; r < 8; ++r) {
                float v = acc[mi][ni][r] + bv;
                size_t idx = (size_t)(row0 + r) * N + col;
                if constexpr (EPI == EP_RES_F32) {
                    ((float*)Cv)[idx] = v + res[idx];
                } else if constexpr (EPI == EP_GELU_BF16) {
                    float g = 0.5f * v * (1.0f + erff(v * 0.70710678118654752f));
                    ((__bf16*)Cv)[idx] = (__bf16)g;
                } else {
                    ((__bf16*)Cv)[idx] = (__bf16)v;
                }
            }
        }
    }
}

// ---------------------------------------------------------------------------
// Row LayerNorm over D=1024: 1 block (256 threads) per row, 4 floats/thread.
// In-place safe (row fully read before writes). Optionally emits bf16 copy.
// ---------------------------------------------------------------------------
__global__ __launch_bounds__(256)
void ln_kernel(const float* __restrict__ x, const float* __restrict__ g,
               const float* __restrict__ b, float* __restrict__ out,
               __bf16* __restrict__ out16, int D)
{
    const int row = blockIdx.x;
    const int tid = threadIdx.x;
    const float4 v = *(const float4*)(x + (size_t)row * D + tid * 4);

    float s  = v.x + v.y + v.z + v.w;
    float ss = v.x * v.x + v.y * v.y + v.z * v.z + v.w * v.w;
#pragma unroll
    for (int off = 16; off > 0; off >>= 1) {
        s  += __shfl_xor(s, off, 32);
        ss += __shfl_xor(ss, off, 32);
    }
    __shared__ float red_s[8], red_ss[8];
    if ((tid & 31) == 0) { red_s[tid >> 5] = s; red_ss[tid >> 5] = ss; }
    __syncthreads();
    s = 0.f; ss = 0.f;
#pragma unroll
    for (int i = 0; i < 8; ++i) { s += red_s[i]; ss += red_ss[i]; }

    const float mu  = s / (float)D;
    const float var = ss / (float)D - mu * mu;
    const float inv = rsqrtf(var + 1e-5f);

    const float4 gg = *(const float4*)(g + tid * 4);
    const float4 bb = *(const float4*)(b + tid * 4);
    float4 o;
    o.x = (v.x - mu) * inv * gg.x + bb.x;
    o.y = (v.y - mu) * inv * gg.y + bb.y;
    o.z = (v.z - mu) * inv * gg.z + bb.z;
    o.w = (v.w - mu) * inv * gg.w + bb.w;
    *(float4*)(out + (size_t)row * D + tid * 4) = o;
    if (out16) {
        bf16x4 h;
        h.x = (__bf16)o.x; h.y = (__bf16)o.y; h.z = (__bf16)o.z; h.w = (__bf16)o.w;
        *(bf16x4*)(out16 + (size_t)row * D + tid * 4) = h;
    }
}

// ---------------------------------------------------------------------------
// Orchestration. seq-len-1 softmax == 1, so MHA == (x_kv @ Wv^T + bv) @ Wo^T + bo.
// ---------------------------------------------------------------------------
extern "C" void kernel_launch(void* const* d_in, const int* in_sizes, int n_in,
                              void* d_out, int out_size, void* d_ws, size_t ws_size,
                              hipStream_t stream)
{
    const int B = 32768, D = 1024, D4 = 4096;
    const size_t BD = (size_t)B * D;

    const float* seq_emb   = (const float*)d_in[0];
    const float* graph_emb = (const float*)d_in[1];
    const float* s2g_in_w  = (const float*)d_in[2];
    const float* s2g_in_b  = (const float*)d_in[3];
    const float* s2g_out_w = (const float*)d_in[4];
    const float* s2g_out_b = (const float*)d_in[5];
    const float* g2s_in_w  = (const float*)d_in[6];
    const float* g2s_in_b  = (const float*)d_in[7];
    const float* g2s_out_w = (const float*)d_in[8];
    const float* g2s_out_b = (const float*)d_in[9];
    const float* seq_w1 = (const float*)d_in[10];
    const float* seq_b1 = (const float*)d_in[11];
    const float* seq_w2 = (const float*)d_in[12];
    const float* seq_b2 = (const float*)d_in[13];
    const float* gr_w1  = (const float*)d_in[14];
    const float* gr_b1  = (const float*)d_in[15];
    const float* gr_w2  = (const float*)d_in[16];
    const float* gr_b2  = (const float*)d_in[17];
    const float* sn1_g = (const float*)d_in[18]; const float* sn1_b = (const float*)d_in[19];
    const float* sn2_g = (const float*)d_in[20]; const float* sn2_b = (const float*)d_in[21];
    const float* gn1_g = (const float*)d_in[22]; const float* gn1_b = (const float*)d_in[23];
    const float* gn2_g = (const float*)d_in[24]; const float* gn2_b = (const float*)d_in[25];

    // ---- workspace carve-out (~872 MB) ----
    char* ws = (char*)d_ws;
    size_t ofs = 0;
    auto carve = [&](size_t bytes) {
        char* p = ws + ofs;
        ofs = (ofs + bytes + 255) & ~(size_t)255;
        return p;
    };
    __bf16* wv_s2g = (__bf16*)carve((size_t)D * D * 2);
    __bf16* wo_s2g = (__bf16*)carve((size_t)D * D * 2);
    __bf16* wv_g2s = (__bf16*)carve((size_t)D * D * 2);
    __bf16* wo_g2s = (__bf16*)carve((size_t)D * D * 2);
    __bf16* w_s1   = (__bf16*)carve((size_t)D4 * D * 2);
    __bf16* w_s2   = (__bf16*)carve((size_t)D4 * D * 2);
    __bf16* w_g1   = (__bf16*)carve((size_t)D4 * D * 2);
    __bf16* w_g2   = (__bf16*)carve((size_t)D4 * D * 2);
    __bf16* S16    = (__bf16*)carve(BD * 2);        // seq_emb bf16
    __bf16* G16    = (__bf16*)carve(BD * 2);        // graph_emb bf16
    __bf16* T16    = (__bf16*)carve(BD * 2);        // V-projection intermediate
    __bf16* H16    = (__bf16*)carve(BD * 4 * 2);    // FFN hidden (post-gelu, bf16)
    __bf16* SRES16 = (__bf16*)carve(BD * 2);        // seq_res bf16
    __bf16* GRES16 = (__bf16*)carve(BD * 2);        // graph_res bf16
    float*  SRESF  = (float*) carve(BD * 4);        // seq_res f32 (also pre-LN acc)
    float*  GRESF  = (float*) carve(BD * 4);        // graph_res f32 (also pre-LN acc)

    float* seq_out   = (float*)d_out;
    float* graph_out = (float*)d_out + BD;

    // ---- conversions: weights (V rows are in_w[2D:3D]) and embeddings ----
    auto cvt = [&](const float* src, __bf16* dst, size_t n) {
        long n4 = (long)(n / 4);
        cvt_f32_bf16<<<(n4 + 255) / 256, 256, 0, stream>>>(src, dst, n4);
    };
    cvt(s2g_in_w + (size_t)2 * D * D, wv_s2g, (size_t)D * D);
    cvt(s2g_out_w,                    wo_s2g, (size_t)D * D);
    cvt(g2s_in_w + (size_t)2 * D * D, wv_g2s, (size_t)D * D);
    cvt(g2s_out_w,                    wo_g2s, (size_t)D * D);
    cvt(seq_w1, w_s1, (size_t)D4 * D);
    cvt(seq_w2, w_s2, (size_t)D4 * D);
    cvt(gr_w1,  w_g1, (size_t)D4 * D);
    cvt(gr_w2,  w_g2, (size_t)D4 * D);
    cvt(seq_emb,   S16, BD);
    cvt(graph_emb, G16, BD);

    const dim3 blk(256);
    const dim3 gD (D  / BN, B / BM);   // N = 1024 -> (4, 256)
    const dim3 gD4(D4 / BN, B / BM);   // N = 4096 -> (16, 256)

    // ===== graph path attention =====
    gemm_bf16_wmma<EP_BF16   ><<<gD, blk, 0, stream>>>(
        S16, wv_s2g, s2g_in_b + 2 * D, nullptr, T16, B, D, D);
    gemm_bf16_wmma<EP_RES_F32><<<gD, blk, 0, stream>>>(
        T16, wo_s2g, s2g_out_b, graph_emb, GRESF, B, D, D);
    ln_kernel<<<B, blk, 0, stream>>>(GRESF, gn1_g, gn1_b, GRESF, GRES16, D);

    // ===== seq path attention =====
    gemm_bf16_wmma<EP_BF16   ><<<gD, blk, 0, stream>>>(
        G16, wv_g2s, g2s_in_b + 2 * D, nullptr, T16, B, D, D);
    gemm_bf16_wmma<EP_RES_F32><<<gD, blk, 0, stream>>>(
        T16, wo_g2s, g2s_out_b, seq_emb, SRESF, B, D, D);
    ln_kernel<<<B, blk, 0, stream>>>(SRESF, sn1_g, sn1_b, SRESF, SRES16, D);

    // ===== seq FFN + LN =====
    gemm_bf16_wmma<EP_GELU_BF16><<<gD4, blk, 0, stream>>>(
        SRES16, w_s1, seq_b1, nullptr, H16, B, D4, D);
    gemm_bf16_wmma<EP_RES_F32  ><<<gD,  blk, 0, stream>>>(
        H16, w_s2, seq_b2, SRESF, SRESF, B, D, D4);       // C aliases res (safe)
    ln_kernel<<<B, blk, 0, stream>>>(SRESF, sn2_g, sn2_b, seq_out, ((__bf16*)nullptr), D);

    // ===== graph FFN + LN =====
    gemm_bf16_wmma<EP_GELU_BF16><<<gD4, blk, 0, stream>>>(
        GRES16, w_g1, gr_b1, nullptr, H16, B, D4, D);
    gemm_bf16_wmma<EP_RES_F32  ><<<gD,  blk, 0, stream>>>(
        H16, w_g2, gr_b2, GRESF, GRESF, B, D, D4);        // C aliases res (safe)
    ln_kernel<<<B, blk, 0, stream>>>(GRESF, gn2_g, gn2_b, graph_out, ((__bf16*)nullptr), D);

    (void)in_sizes; (void)n_in; (void)out_size; (void)ws_size;
}
